// ANIModel_42691974922491
// MI455X (gfx1250) — compile-verified, matrix operand order checked by reading, not run on hardware
//
#include <hip/hip_runtime.h>

// ---------------- problem constants (match reference) ----------------
#define NSPEC   4
#define NCONF   2048
#define NATOMA  64
#define NATOMS  (NCONF * NATOMA)   // 131072
#define DD      384
#define HH1     160
#define HH2     128
#define HH3     96
#define CELU_A  0.1f

#define KT1 12   // DD/32
#define JT1 10   // HH1/16
#define JT1H 5   // half of JT1
#define KT2 5    // HH1/32
#define JT2 8    // HH2/16
#define KT3 4    // HH2/32
#define JT3 6    // HH3/16

#define WG_PER_SPEC 64
#define BLOCK_THREADS 512          // 16 wave32s
#define WAVES_PER_BLOCK 16

typedef __attribute__((ext_vector_type(16))) __bf16 v16bf;
typedef __attribute__((ext_vector_type(8)))  float  v8f;

// ---------------- LDS carve (bytes) ----------------
#define OFF_W1P   0u
#define OFF_W2P   (OFF_W1P + (unsigned)(KT1*JT1*32*16*2))   // 122880
#define OFF_W3P   (OFF_W2P + (unsigned)(KT2*JT2*32*16*2))   // +40960 = 163840
#define OFF_STG   (OFF_W3P + (unsigned)(KT3*JT3*32*16*2))   // +24576 = 188416
#define OFF_B1    (OFF_STG + (unsigned)(WAVES_PER_BLOCK*16*HH1*2)) // +81920 = 270336
#define OFF_B2    (OFF_B1 + HH1*4)
#define OFF_B3    (OFF_B2 + HH2*4)
#define OFF_W4    (OFF_B3 + HH3*4)
#define OFF_B4    (OFF_W4 + HH3*4)
#define SMEM_BYTES ((OFF_B4 + 4 + 31u) & ~31u)              // 272288

__device__ __forceinline__ float celu_f(float x) {
    return x > 0.0f ? x : (CELU_A * (__expf(x * (1.0f / CELU_A)) - 1.0f));
}

__device__ __forceinline__ v16bf load_a16_global(const float* __restrict__ p) {
    const float4* q = (const float4*)p;          // 64B-aligned (row stride 1536B, off mult of 64)
    float4 f0 = q[0], f1 = q[1], f2 = q[2], f3 = q[3];
    v16bf a;
    a[0]=(__bf16)f0.x;  a[1]=(__bf16)f0.y;  a[2]=(__bf16)f0.z;  a[3]=(__bf16)f0.w;
    a[4]=(__bf16)f1.x;  a[5]=(__bf16)f1.y;  a[6]=(__bf16)f1.z;  a[7]=(__bf16)f1.w;
    a[8]=(__bf16)f2.x;  a[9]=(__bf16)f2.y;  a[10]=(__bf16)f2.z; a[11]=(__bf16)f2.w;
    a[12]=(__bf16)f3.x; a[13]=(__bf16)f3.y; a[14]=(__bf16)f3.z; a[15]=(__bf16)f3.w;
    return a;
}

#define WMMA_BF16(A, B, C) \
    __builtin_amdgcn_wmma_f32_16x16x32_bf16(false, (A), false, (B), (short)0, (C), false, false)

// ---------------- pass 1: zero species counters ----------------
__global__ void zero_counts_kernel(int* counts) {
    if (threadIdx.x < NSPEC) counts[threadIdx.x] = 0;
}

// ---------------- pass 2: bucket atoms by species ----------------
__global__ void scatter_kernel(const int* __restrict__ species,
                               int* __restrict__ counts,
                               int* __restrict__ perm) {
    int i = blockIdx.x * blockDim.x + threadIdx.x;
    if (i < NATOMS) {
        int s = species[i];
        int pos = atomicAdd(&counts[s], 1);
        perm[s * NATOMS + pos] = i;   // order nondeterministic; results order-invariant
    }
}

// ---------------- pass 3: species-batched WMMA MLP ----------------
__global__ __launch_bounds__(BLOCK_THREADS, 1)
void ani_mlp_kernel(const float* __restrict__ aev,
                    const float* __restrict__ W1, const float* __restrict__ b1,
                    const float* __restrict__ W2, const float* __restrict__ b2,
                    const float* __restrict__ W3, const float* __restrict__ b3,
                    const float* __restrict__ W4, const float* __restrict__ b4,
                    const int*   __restrict__ counts,
                    const int*   __restrict__ perm,
                    float*       __restrict__ per_atom) {
    extern __shared__ char smem[];
    __bf16* W1p = (__bf16*)(smem + OFF_W1P);
    __bf16* W2p = (__bf16*)(smem + OFF_W2P);
    __bf16* W3p = (__bf16*)(smem + OFF_W3P);
    __bf16* stg = (__bf16*)(smem + OFF_STG);
    float*  b1s = (float*)(smem + OFF_B1);
    float*  b2s = (float*)(smem + OFF_B2);
    float*  b3s = (float*)(smem + OFF_B3);
    float*  w4s = (float*)(smem + OFF_W4);
    float*  b4s = (float*)(smem + OFF_B4);

    const int tid = threadIdx.x;
    const int s   = blockIdx.x & (NSPEC - 1);
    const int wg  = blockIdx.x >> 2;

    // ---- cooperative weight pack: fp32 global -> bf16 LDS in B-fragment layout ----
    // fragment f = kt*JT + jt; lane fl owns 16 contiguous bf16 = column n, 16 k's
    {
        const float* Wg = W1 + (size_t)s * DD * HH1;
        for (int u = tid; u < KT1 * JT1 * 32; u += BLOCK_THREADS) {
            int fl = u & 31, f = u >> 5;
            int jt = f % JT1, kt = f / JT1;
            int n  = jt * 16 + (fl & 15);
            int k0 = kt * 32 + (fl >> 4) * 16;
            __bf16* dst = W1p + (size_t)u * 16;
            #pragma unroll
            for (int i = 0; i < 16; ++i) dst[i] = (__bf16)Wg[(k0 + i) * HH1 + n];
        }
    }
    {
        const float* Wg = W2 + (size_t)s * HH1 * HH2;
        for (int u = tid; u < KT2 * JT2 * 32; u += BLOCK_THREADS) {
            int fl = u & 31, f = u >> 5;
            int jt = f & (JT2 - 1), kt = f >> 3;
            int n  = jt * 16 + (fl & 15);
            int k0 = kt * 32 + (fl >> 4) * 16;
            __bf16* dst = W2p + (size_t)u * 16;
            #pragma unroll
            for (int i = 0; i < 16; ++i) dst[i] = (__bf16)Wg[(k0 + i) * HH2 + n];
        }
    }
    {
        const float* Wg = W3 + (size_t)s * HH2 * HH3;
        for (int u = tid; u < KT3 * JT3 * 32; u += BLOCK_THREADS) {
            int fl = u & 31, f = u >> 5;
            int jt = f % JT3, kt = f / JT3;
            int n  = jt * 16 + (fl & 15);
            int k0 = kt * 32 + (fl >> 4) * 16;
            __bf16* dst = W3p + (size_t)u * 16;
            #pragma unroll
            for (int i = 0; i < 16; ++i) dst[i] = (__bf16)Wg[(k0 + i) * HH3 + n];
        }
    }
    for (int i = tid; i < HH1; i += BLOCK_THREADS) b1s[i] = b1[s * HH1 + i];
    for (int i = tid; i < HH2; i += BLOCK_THREADS) b2s[i] = b2[s * HH2 + i];
    for (int i = tid; i < HH3; i += BLOCK_THREADS) {
        b3s[i] = b3[s * HH3 + i];
        w4s[i] = W4[s * HH3 + i];
    }
    if (tid == 0) b4s[0] = b4[s];
    __syncthreads();

    const int lane = tid & 31;
    const int wid  = tid >> 5;
    const int lr   = lane & 15;   // A-row / C-column within tile
    const int lh   = lane >> 4;   // half-wave select
    __bf16* st = stg + wid * (16 * HH1);

    const int   cnt    = counts[s];
    const int   ntiles = (cnt + 15) >> 4;
    const float b4v    = b4s[0];

    for (int t = wg * WAVES_PER_BLOCK + wid; t < ntiles; t += WG_PER_SPEC * WAVES_PER_BLOCK) {
        int r = t * 16 + lr;
        int atom = perm[s * NATOMS + min(r, cnt - 1)];   // clamp pad rows to a valid atom
        const float* xrow = aev + (size_t)atom * DD;

        // ---------- layer 1: [16x384] x [384x160] ----------
        // Preload ALL 12 A-fragments while accumulators are dead, so the
        // compiler can clause the 48 global_load_b128s and hide HBM latency.
        v16bf af[KT1];
        #pragma unroll
        for (int kt = 0; kt < KT1; ++kt)
            af[kt] = load_a16_global(xrow + kt * 32 + lh * 16);

        // Two output-column halves (5 accum tiles each) to cap VGPR pressure
        // at ~160 so the scheduler can double-buffer B-fragment ds_loads.
        #pragma unroll
        for (int g = 0; g < 2; ++g) {
            v8f c1[JT1H];
            #pragma unroll
            for (int j = 0; j < JT1H; ++j) { v8f z = {0,0,0,0,0,0,0,0}; c1[j] = z; }
            #pragma unroll
            for (int kt = 0; kt < KT1; ++kt) {
                #pragma unroll
                for (int j = 0; j < JT1H; ++j) {
                    int jt = g * JT1H + j;
                    v16bf bfr = *(const v16bf*)(W1p + ((kt * JT1 + jt) * 32 + lane) * 16);
                    c1[j] = WMMA_BF16(af[kt], bfr, c1[j]);
                }
            }
            // bias + CELU, stage bf16 activations (cols g*80 .. g*80+79)
            #pragma unroll
            for (int j = 0; j < JT1H; ++j) {
                int jt = g * JT1H + j;
                float bias = b1s[jt * 16 + lr];
                #pragma unroll
                for (int v = 0; v < 8; ++v) {
                    float h = celu_f(c1[j][v] + bias);
                    st[(v + lh * 8) * HH1 + jt * 16 + lr] = (__bf16)h;
                }
            }
        }
        asm volatile("s_wait_dscnt 0" ::: "memory");   // cross-lane LDS handoff

        // ---------- layer 2: [16x160] x [160x128] ----------
        v8f c2[JT2];
        #pragma unroll
        for (int j = 0; j < JT2; ++j) { v8f z = {0,0,0,0,0,0,0,0}; c2[j] = z; }
        #pragma unroll
        for (int kt = 0; kt < KT2; ++kt) {
            v16bf a = *(const v16bf*)(st + lr * HH1 + kt * 32 + lh * 16);
            #pragma unroll
            for (int jt = 0; jt < JT2; ++jt) {
                v16bf bfr = *(const v16bf*)(W2p + ((kt * JT2 + jt) * 32 + lane) * 16);
                c2[jt] = WMMA_BF16(a, bfr, c2[jt]);
            }
        }
        #pragma unroll
        for (int jt = 0; jt < JT2; ++jt) {
            float bias = b2s[jt * 16 + lr];
            #pragma unroll
            for (int v = 0; v < 8; ++v) {
                float h = celu_f(c2[jt][v] + bias);
                st[(v + lh * 8) * HH2 + jt * 16 + lr] = (__bf16)h;
            }
        }
        asm volatile("s_wait_dscnt 0" ::: "memory");

        // ---------- layer 3: [16x128] x [128x96] ----------
        v8f c3[JT3];
        #pragma unroll
        for (int j = 0; j < JT3; ++j) { v8f z = {0,0,0,0,0,0,0,0}; c3[j] = z; }
        #pragma unroll
        for (int kt = 0; kt < KT3; ++kt) {
            v16bf a = *(const v16bf*)(st + lr * HH2 + kt * 32 + lh * 16);
            #pragma unroll
            for (int jt = 0; jt < JT3; ++jt) {
                v16bf bfr = *(const v16bf*)(W3p + ((kt * JT3 + jt) * 32 + lane) * 16);
                c3[jt] = WMMA_BF16(a, bfr, c3[jt]);
            }
        }

        // ---------- layer 4: [16x96] . w4[96] -> y[16]  (VALU + shfl reduce) ----------
        float acc[8];
        #pragma unroll
        for (int v = 0; v < 8; ++v) acc[v] = 0.0f;
        #pragma unroll
        for (int jt = 0; jt < JT3; ++jt) {
            float bias = b3s[jt * 16 + lr];
            float wv   = w4s[jt * 16 + lr];
            #pragma unroll
            for (int v = 0; v < 8; ++v)
                acc[v] += celu_f(c3[jt][v] + bias) * wv;
        }
        #pragma unroll
        for (int v = 0; v < 8; ++v) {
            float a0 = acc[v];
            a0 += __shfl_xor(a0, 1, 16);
            a0 += __shfl_xor(a0, 2, 16);
            a0 += __shfl_xor(a0, 4, 16);
            a0 += __shfl_xor(a0, 8, 16);
            acc[v] = a0;
        }
        // lanes 0..7 hold y[0..7], lanes 16..23 hold y[8..15]
        float y = 0.0f;
        #pragma unroll
        for (int v = 0; v < 8; ++v) y = (lr == v) ? acc[v] : y;
        int m = lh * 8 + lr;                // valid only when lr < 8
        int rOut = t * 16 + m;
        if (lr < 8 && rOut < cnt) {
            int atomOut = perm[s * NATOMS + rOut];
            per_atom[atomOut] = y + b4v;
        }
    }
}

// ---------------- pass 4: per-conformation reduction ----------------
__global__ void reduce_energy_kernel(const float* __restrict__ per_atom,
                                     float* __restrict__ energies) {
    int lane = threadIdx.x & 31;
    int wid  = threadIdx.x >> 5;
    int c = blockIdx.x * (blockDim.x >> 5) + wid;   // one wave per conformation
    if (c >= NCONF) return;
    float v = per_atom[c * NATOMA + lane] + per_atom[c * NATOMA + 32 + lane];
    v += __shfl_xor(v, 16, 32);
    v += __shfl_xor(v, 8, 32);
    v += __shfl_xor(v, 4, 32);
    v += __shfl_xor(v, 2, 32);
    v += __shfl_xor(v, 1, 32);
    if (lane == 0) energies[c] = v;
}

// ---------------- host launcher ----------------
extern "C" void kernel_launch(void* const* d_in, const int* in_sizes, int n_in,
                              void* d_out, int out_size, void* d_ws, size_t ws_size,
                              hipStream_t stream) {
    const int*   species = (const int*)d_in[0];
    const float* aev     = (const float*)d_in[1];
    const float* W1 = (const float*)d_in[2];
    const float* b1 = (const float*)d_in[3];
    const float* W2 = (const float*)d_in[4];
    const float* b2 = (const float*)d_in[5];
    const float* W3 = (const float*)d_in[6];
    const float* b3 = (const float*)d_in[7];
    const float* W4 = (const float*)d_in[8];
    const float* b4 = (const float*)d_in[9];

    // workspace carve
    char* ws = (char*)d_ws;
    int*   counts   = (int*)ws;                                        // 16 B (pad to 256)
    int*   perm     = (int*)(ws + 256);                                // 4*N*4 = 2 MB
    float* per_atom = (float*)(ws + 256 + (size_t)NSPEC * NATOMS * 4); // N*4

    // output: [species ints (N)] then [energies floats (NCONF)]
    hipMemcpyAsync(d_out, (const void*)species, (size_t)NATOMS * sizeof(int),
                   hipMemcpyDeviceToDevice, stream);
    float* energies = (float*)d_out + NATOMS;

    zero_counts_kernel<<<1, 32, 0, stream>>>(counts);
    scatter_kernel<<<NATOMS / 256, 256, 0, stream>>>(species, counts, perm);

    (void)hipFuncSetAttribute((const void*)ani_mlp_kernel,
                              hipFuncAttributeMaxDynamicSharedMemorySize,
                              (int)SMEM_BYTES);
    ani_mlp_kernel<<<NSPEC * WG_PER_SPEC, BLOCK_THREADS, SMEM_BYTES, stream>>>(
        aev, W1, b1, W2, b2, W3, b3, W4, b4, counts, perm, per_atom);

    reduce_energy_kernel<<<NCONF / 8, 256, 0, stream>>>(per_atom, energies);
}
